// GCNConv0TPK_523986010689
// MI455X (gfx1250) — matrix operand order repaired and checked by reading, not compiled, not run on hardware
//
#include <hip/hip_runtime.h>
#include <math.h>

typedef __attribute__((ext_vector_type(16))) _Float16 v16h;
typedef __attribute__((ext_vector_type(8)))  float    v8f;

#define FDIM 128
#define NGRAPH 512

union FragH { v16h h; unsigned int u[8]; };

// ---------------- degree / normalization ----------------
__global__ void k_init_deg(float* deg, int N) {
  int i = blockIdx.x * blockDim.x + threadIdx.x;
  if (i < N) deg[i] = 1.0f;                 // self-loop
}
__global__ void k_count_deg(const int* __restrict__ ei, int E, float* deg) {
  int e = blockIdx.x * blockDim.x + threadIdx.x;
  if (e < E) atomicAdd(&deg[ei[E + e]], 1.0f);   // col = target
}
__global__ void k_rsqrt_inplace(float* deg, int N) {
  int i = blockIdx.x * blockDim.x + threadIdx.x;
  if (i < N) deg[i] = rsqrtf(deg[i]);       // deg >= 1 always
}

// ---------------- precision conversion ----------------
__global__ void k_f32_to_f16(const float* __restrict__ src, _Float16* __restrict__ dst, int n) {
  int i = blockIdx.x * blockDim.x + threadIdx.x;
  if (i < n) dst[i] = (_Float16)src[i];
}
// W is [K=in][N=out] row-major; store transposed [N][K] in f16 so B-fragment
// loads (consecutive K pairs for a fixed output column) are contiguous 32-bit loads.
__global__ void k_transpose_w16(const float* __restrict__ W, _Float16* __restrict__ Wt) {
  int i = blockIdx.x * blockDim.x + threadIdx.x;
  if (i < FDIM * FDIM) {
    int k = i / FDIM, n = i % FDIM;
    Wt[n * FDIM + k] = (_Float16)W[k * FDIM + n];
  }
}

// ---------------- WMMA GEMM ----------------
// xs  = (Hin @ W) * dinv[row]          (source-normalized messages)
// agg = xs * dinv[row]                 (self-loop term folded in; initializes agg)
// One wave per 16 node rows, producing all 128 output columns (8 N-tiles).
// Requires N % 16 == 0 (true: 100000). Wave-uniform guard keeps EXEC all-1s for WMMA.
__global__ __launch_bounds__(256)
void k_gemm_gcn(const _Float16* __restrict__ Hin, const _Float16* __restrict__ Wt,
                const float* __restrict__ dinv, float* __restrict__ xs,
                float* __restrict__ agg, int N) {
  const int lane = threadIdx.x & 31;
  const int wave = threadIdx.x >> 5;
  const int half = lane >> 4;      // lane group 0/1
  const int m    = lane & 15;      // M (A) / N (B,C) index within tile
  const int rowBase = (blockIdx.x * 8 + wave) * 16;
  if (rowBase + 16 > N) return;    // uniform per wave

  v8f acc[8] = {};
  const _Float16* arow = Hin + (long)(rowBase + m) * FDIM;

  #pragma unroll
  for (int kk = 0; kk < 4; ++kk) {           // K = 128 in 4 chunks of 32
    FragH a;
    #pragma unroll
    for (int v = 0; v < 8; ++v) {
      // 16-bit A 16x32 layout: VGPR v holds K = 2*(v&3) + 16*(v>>2) + 8*half (+1)
      int k0 = kk * 32 + 2 * (v & 3) + 16 * (v >> 2) + 8 * half;
      a.u[v] = *(const unsigned int*)(arow + k0);
    }
    #pragma unroll
    for (int nt = 0; nt < 8; ++nt) {
      FragH b;
      const _Float16* wcol = Wt + (long)(nt * 16 + m) * FDIM;
      #pragma unroll
      for (int v = 0; v < 8; ++v) {
        // 16-bit B 32x16 layout: VGPR v holds K = 2*(v&3) + 16*(v>>2) + 8*half (+1), N = lane&15
        int k0 = kk * 32 + 2 * (v & 3) + 16 * (v >> 2) + 8 * half;
        b.u[v] = *(const unsigned int*)(wcol + k0);
      }
      acc[nt] = __builtin_amdgcn_wmma_f32_16x16x32_f16(
          false, a.h, false, b.h, (short)0, acc[nt], false, false);
    }
  }

  // Epilogue: C/D layout -> row = rowBase + r + 8*half, col = nt*16 + (lane&15).
  #pragma unroll
  for (int r = 0; r < 8; ++r) {
    int row = rowBase + r + 8 * half;
    float dv = dinv[row];
    long base = (long)row * FDIM;
    #pragma unroll
    for (int nt = 0; nt < 8; ++nt) {
      int col = nt * 16 + m;
      float sv = acc[nt][r] * dv;     // pre-scaled by source norm
      xs[base + col]  = sv;
      agg[base + col] = sv * dv;      // self-loop: xw * dinv^2
    }
  }
}

// ---------------- edge scatter: agg[col] += xs[row] * dinv[col] ----------------
// One wave per edge; each lane handles 4 contiguous features (float4, 16B aligned).
// Edge index is wave-uniform: force it through readfirstlane so the backend can
// scalarize the index / norm loads (s_load) instead of 32 replicated VMEM loads.
__global__ void k_edge_scatter(const int* __restrict__ ei, int E,
                               const float* __restrict__ xs,
                               const float* __restrict__ dinv,
                               float* __restrict__ agg) {
  long t = (long)blockIdx.x * blockDim.x + threadIdx.x;
  long e64 = t >> 5;
  if (e64 >= E) return;
  int e = __builtin_amdgcn_readfirstlane((int)e64);   // wave-uniform
  int ch = ((int)t & 31) * 4;
  int row = ei[e];
  int col = ei[E + e];
  float dcol = dinv[col];
  const float4 v = *(const float4*)(xs + (long)row * FDIM + ch);
  float* dst = agg + (long)col * FDIM + ch;
  atomicAdd(dst + 0, v.x * dcol);
  atomicAdd(dst + 1, v.y * dcol);
  atomicAdd(dst + 2, v.z * dcol);
  atomicAdd(dst + 3, v.w * dcol);
}

// ---------------- bias + relu, emit f16 features for next layer ----------------
__global__ void k_bias_relu_h(const float* __restrict__ agg, const float* __restrict__ bias,
                              _Float16* __restrict__ Hout, int n) {
  int i = blockIdx.x * blockDim.x + threadIdx.x;
  if (i < n) {
    float v = agg[i] + bias[i & (FDIM - 1)];
    Hout[i] = (_Float16)(v > 0.0f ? v : 0.0f);
  }
}

// ---------------- global mean pool ----------------
__global__ void k_zero_pool(float* pool, float* cnt) {
  int i = blockIdx.x * blockDim.x + threadIdx.x;
  if (i < NGRAPH * FDIM) pool[i] = 0.0f;
  if (i < NGRAPH) cnt[i] = 0.0f;
}
__global__ void k_pool(const _Float16* __restrict__ H, const int* __restrict__ batch,
                       float* __restrict__ pool, float* __restrict__ cnt, int N) {
  long t = (long)blockIdx.x * blockDim.x + threadIdx.x;
  long v64 = t >> 5;
  if (v64 >= N) return;
  int v = __builtin_amdgcn_readfirstlane((int)v64);   // wave-uniform node id
  int ch = ((int)t & 31) * 4;
  int b = batch[v];
  const _Float16* src = H + (long)v * FDIM + ch;
  float* dst = pool + (long)b * FDIM + ch;
  atomicAdd(dst + 0, (float)src[0]);
  atomicAdd(dst + 1, (float)src[1]);
  atomicAdd(dst + 2, (float)src[2]);
  atomicAdd(dst + 3, (float)src[3]);
  if (ch == 0) atomicAdd(&cnt[b], 1.0f);
}

// ---------------- MLP head + log_softmax (tiny: 512 x ~4 MFLOP) ----------------
__global__ __launch_bounds__(64)
void k_mlp_head(const float* __restrict__ pool, const float* __restrict__ cnt,
                const float* __restrict__ Wl1, const float* __restrict__ bl1,
                const float* __restrict__ Wl2, const float* __restrict__ bl2,
                float* __restrict__ out) {
  __shared__ float g[FDIM];
  __shared__ float hid[64];
  __shared__ float logit[10];
  int gr = blockIdx.x;
  int t = threadIdx.x;
  float c = cnt[gr];
  c = c > 1.0f ? c : 1.0f;
  float inv = 1.0f / c;
  g[t]      = pool[gr * FDIM + t]      * inv;
  g[t + 64] = pool[gr * FDIM + t + 64] * inv;
  __syncthreads();
  float acc = bl1[t];
  #pragma unroll 4
  for (int k = 0; k < FDIM; ++k) acc += g[k] * Wl1[k * 64 + t];
  hid[t] = acc > 0.0f ? acc : 0.0f;
  __syncthreads();
  if (t < 10) {
    float a = bl2[t];
    for (int k = 0; k < 64; ++k) a += hid[k] * Wl2[k * 10 + t];
    logit[t] = a;
  }
  __syncthreads();
  if (t == 0) {
    float mx = logit[0];
    for (int i = 1; i < 10; ++i) mx = fmaxf(mx, logit[i]);
    float s = 0.0f;
    for (int i = 0; i < 10; ++i) s += expf(logit[i] - mx);
    float lse = mx + logf(s);
    for (int i = 0; i < 10; ++i) out[gr * 10 + i] = logit[i] - lse;
  }
}

extern "C" void kernel_launch(void* const* d_in, const int* in_sizes, int n_in,
                              void* d_out, int out_size, void* d_ws, size_t ws_size,
                              hipStream_t stream) {
  const float* x    = (const float*)d_in[0];
  const int*   ei   = (const int*)d_in[1];
  const int*   batch= (const int*)d_in[2];
  const float* W1   = (const float*)d_in[3];  const float* b1  = (const float*)d_in[4];
  const float* W2   = (const float*)d_in[5];  const float* b2  = (const float*)d_in[6];
  const float* W3   = (const float*)d_in[7];  const float* b3  = (const float*)d_in[8];
  const float* Wl1  = (const float*)d_in[9];  const float* bl1 = (const float*)d_in[10];
  const float* Wl2  = (const float*)d_in[11]; const float* bl2 = (const float*)d_in[12];
  float* out = (float*)d_out;

  const int N = in_sizes[0] / FDIM;   // 100000
  const int E = in_sizes[1] / 2;      // 1600000
  const int NF = N * FDIM;

  // workspace layout (256B aligned slabs)
  char* ws = (char*)d_ws;
  size_t off = 0;
  auto alloc = [&](size_t bytes) -> void* {
    void* p = ws + off;
    off += (bytes + 255) & ~(size_t)255;
    return p;
  };
  _Float16* Hh   = (_Float16*)alloc((size_t)NF * 2);          // f16 features (reused in place)
  float*    xs   = (float*)   alloc((size_t)NF * 4);          // (X @ W) * dinv[row]
  float*    agg  = (float*)   alloc((size_t)NF * 4);          // aggregated messages
  float*    dinv = (float*)   alloc((size_t)N * 4);           // deg -> rsqrt(deg), in place
  _Float16* Wt1  = (_Float16*)alloc((size_t)FDIM * FDIM * 2);
  _Float16* Wt2  = (_Float16*)alloc((size_t)FDIM * FDIM * 2);
  _Float16* Wt3  = (_Float16*)alloc((size_t)FDIM * FDIM * 2);
  float*    pool = (float*)   alloc((size_t)NGRAPH * FDIM * 4);
  float*    cnt  = (float*)   alloc((size_t)NGRAPH * 4);

  const int TB = 256;
  int gN   = (N + TB - 1) / TB;
  int gE   = (E + TB - 1) / TB;
  int gNF  = (NF + TB - 1) / TB;
  int gW   = (FDIM * FDIM + TB - 1) / TB;
  int gGemm = ((N / 16) + 7) / 8;                       // 8 waves per block, 16 rows per wave
  long eThreads = (long)E * 32;
  int gEdge = (int)((eThreads + TB - 1) / TB);
  long pThreads = (long)N * 32;
  int gPool = (int)((pThreads + TB - 1) / TB);
  int gZero = (NGRAPH * FDIM + TB - 1) / TB;

  // normalization
  k_init_deg<<<gN, TB, 0, stream>>>(dinv, N);
  k_count_deg<<<gE, TB, 0, stream>>>(ei, E, dinv);
  k_rsqrt_inplace<<<gN, TB, 0, stream>>>(dinv, N);

  // precision prep
  k_f32_to_f16<<<gNF, TB, 0, stream>>>(x, Hh, NF);
  k_transpose_w16<<<gW, TB, 0, stream>>>(W1, Wt1);
  k_transpose_w16<<<gW, TB, 0, stream>>>(W2, Wt2);
  k_transpose_w16<<<gW, TB, 0, stream>>>(W3, Wt3);
  k_zero_pool<<<gZero, TB, 0, stream>>>(pool, cnt);

  // three GCN layers (Hh is overwritten in place between layers)
  const _Float16* Wts[3]  = {Wt1, Wt2, Wt3};
  const float*    bias[3] = {b1, b2, b3};
  for (int l = 0; l < 3; ++l) {
    k_gemm_gcn<<<gGemm, TB, 0, stream>>>(Hh, Wts[l], dinv, xs, agg, N);
    k_edge_scatter<<<gEdge, TB, 0, stream>>>(ei, E, xs, dinv, agg);
    k_bias_relu_h<<<gNF, TB, 0, stream>>>(agg, bias[l], Hh, NF);
  }

  // pooling + head
  k_pool<<<gPool, TB, 0, stream>>>(Hh, batch, pool, cnt, N);
  k_mlp_head<<<NGRAPH, 64, 0, stream>>>(pool, cnt, Wl1, bl1, Wl2, bl2, out);
}
